// SolvGNN_37778532335669
// MI455X (gfx1250) — matrix operand order repaired and checked by reading, not compiled, not run on hardware
//
#include <hip/hip_runtime.h>
#include <stdint.h>

// ---------------- problem constants ----------------
#define NNODES 4096
#define NEDGES 4096
#define NGRAPH 128
#define FIN    74
#define KP1    96        // 74 padded to multiple of 32
#define DD     256
#define EHH    512
#define KBIG   (EHH*DD)  // 131072

typedef __bf16 bf16;
typedef bf16  v16bf __attribute__((ext_vector_type(16)));
typedef bf16  v8bf  __attribute__((ext_vector_type(8)));
typedef float v8f   __attribute__((ext_vector_type(8)));

union V16 { uint4 q[2]; v8bf h[2]; v16bf v; };

// ---------------- small utility kernels ----------------
__global__ void k_deg(const int* __restrict__ src, const int* __restrict__ dst,
                      float* __restrict__ outDeg, float* __restrict__ inDeg) {
    long i = blockIdx.x * 256L + threadIdx.x;
    if (i >= NEDGES) return;
    atomicAdd(&outDeg[src[i]], 1.f);
    atomicAdd(&inDeg[dst[i]], 1.f);
}

__global__ void k_invdeg(const float* __restrict__ outDeg, const float* __restrict__ inDeg,
                         float* __restrict__ inv_s, float* __restrict__ inv_d) {
    long i = blockIdx.x * 256L + threadIdx.x;
    if (i >= NNODES) return;
    inv_s[i] = rsqrtf(fmaxf(outDeg[i], 1.f));
    inv_d[i] = rsqrtf(fmaxf(inDeg[i], 1.f));
}

// fp32 -> bf16 convert, optional per-row scale (row = i / K)
__global__ void k_cvt(const float* __restrict__ src, bf16* __restrict__ dst,
                      const float* __restrict__ scale, int K, long total) {
    long i = blockIdx.x * 256L + threadIdx.x;
    if (i >= total) return;
    float s = scale ? scale[i / K] : 1.f;
    dst[i] = (bf16)(src[i] * s);
}

// fp32 [K,Nc] -> bf16 transposed+padded [Nc,Kp]
__global__ void k_cvtT(const float* __restrict__ src, bf16* __restrict__ dst,
                       int K, int Nc, int Kp, long total) {
    long i = blockIdx.x * 256L + threadIdx.x;
    if (i >= total) return;
    int n = (int)(i / Kp), k = (int)(i % Kp);
    float v = (k < K) ? src[(size_t)k * Nc + n] : 0.f;
    dst[i] = (bf16)v;
}

// he[e,k] = leaky(edge_type[e]*We1[k] + be1[k]) in bf16
__global__ void k_heb(const float* __restrict__ et, const float* __restrict__ We1,
                      const float* __restrict__ be1, bf16* __restrict__ heb) {
    long i = blockIdx.x * 256L + threadIdx.x;
    if (i >= (long)NEDGES * EHH) return;
    int e = (int)(i >> 9), k = (int)(i & 511);
    float v = et[e] * We1[k] + be1[k];
    v = v > 0.f ? v : 0.01f * v;
    heb[i] = (bf16)v;
}

__global__ void k_scatter1(const float* __restrict__ x, const int* __restrict__ src,
                           const int* __restrict__ dst, const float* __restrict__ inv_s,
                           float* __restrict__ agg) {
    long i = blockIdx.x * 256L + threadIdx.x;
    if (i >= (long)NEDGES * FIN) return;
    int e = (int)(i / FIN), f = (int)(i % FIN);
    int s = src[e], d = dst[e];
    atomicAdd(&agg[(size_t)d * KP1 + f], x[(size_t)s * FIN + f] * inv_s[s]);
}

__global__ void k_scatter2(const float* __restrict__ h, const int* __restrict__ src,
                           const int* __restrict__ dst, const float* __restrict__ inv_s,
                           float* __restrict__ agg) {
    long i = blockIdx.x * 256L + threadIdx.x;
    if (i >= (long)NEDGES * DD) return;
    int e = (int)(i >> 8), f = (int)(i & 255);
    int s = src[e], d = dst[e];
    atomicAdd(&agg[(size_t)d * DD + f], h[(size_t)s * DD + f] * inv_s[s]);
}

__global__ void k_gather_src(const float* __restrict__ h, const int* __restrict__ src,
                             bf16* __restrict__ hsb) {
    long i = blockIdx.x * 256L + threadIdx.x;
    if (i >= (long)NEDGES * DD) return;
    int e = (int)(i >> 8), f = (int)(i & 255);
    hsb[i] = (bf16)h[(size_t)src[e] * DD + f];
}

__global__ void k_scatter_msg(const float* __restrict__ msg, const int* __restrict__ dst,
                              float* __restrict__ mbuf) {
    long i = blockIdx.x * 256L + threadIdx.x;
    if (i >= (long)NEDGES * DD) return;
    int e = (int)(i >> 8), f = (int)(i & 255);
    atomicAdd(&mbuf[(size_t)dst[e] * DD + f], msg[i]);
}

__global__ void k_msg_act(const float* __restrict__ mbuf, const float* __restrict__ b_nn,
                          bf16* __restrict__ mb) {
    long i = blockIdx.x * 256L + threadIdx.x;
    if (i >= (long)NNODES * DD) return;
    int f = (int)(i & 255);
    mb[i] = (bf16)fmaxf(mbuf[i] + b_nn[f], 0.f);
}

__global__ void k_gru(const float* __restrict__ gi, const float* __restrict__ gh,
                      const float* __restrict__ hidden, float* __restrict__ hout) {
    long i = blockIdx.x * 256L + threadIdx.x;
    if (i >= (long)NNODES * DD) return;
    int v = (int)(i >> 8), f = (int)(i & 255);
    const float* giv = gi + (size_t)v * 768;
    const float* ghv = gh + (size_t)v * 768;
    float r = 1.f / (1.f + expf(-(giv[f]       + ghv[f])));
    float z = 1.f / (1.f + expf(-(giv[256 + f] + ghv[256 + f])));
    float n = tanhf(giv[512 + f] + r * ghv[512 + f]);
    hout[i] = (1.f - z) * n + z * hidden[i];
}

__global__ void k_node_read(const float* __restrict__ h, const int* __restrict__ gid,
                            float* __restrict__ nodesum, float* __restrict__ ncnt) {
    long i = blockIdx.x * 256L + threadIdx.x;
    if (i >= (long)NNODES * DD) return;
    int v = (int)(i >> 8), f = (int)(i & 255);
    int g = gid[v];
    atomicAdd(&nodesum[(size_t)g * DD + f], h[i]);
    if (f == 0) atomicAdd(&ncnt[g], 1.f);
}

__global__ void k_edge_read(const float* __restrict__ et, const int* __restrict__ gid,
                            float* __restrict__ edgesum, float* __restrict__ ecnt) {
    long i = blockIdx.x * 256L + threadIdx.x;
    if (i >= NEDGES) return;
    int g = gid[i];
    atomicAdd(&edgesum[g], et[i]);
    atomicAdd(&ecnt[g], 1.f);
}

// ---------------- generic bf16 WMMA GEMM ----------------
// C[M,N] = act( A[M,Kp](bf16) @ Bt[N,Kp]^T (bf16) + bias + (acc? C : 0) )
__global__ void k_gemm(const bf16* __restrict__ A, const bf16* __restrict__ Bt,
                       float* __restrict__ C, int M, int N, int Kp,
                       const float* __restrict__ bias, int act, int acc) {
    int w = blockIdx.x * (blockDim.x >> 5) + (threadIdx.x >> 5);
    int ntN = N >> 4;
    if (w >= (M >> 4) * ntN) return;
    int tn = w % ntN, tm = w / ntN;
    int lane = threadIdx.x & 31, half = lane >> 4, lrow = lane & 15;

    const bf16* Ar = A + (size_t)((tm << 4) + lrow) * Kp + half * 8;   // A: interleaved K layout
    const bf16* Br = Bt + (size_t)((tn << 4) + lrow) * Kp + half * 8;  // B: interleaved K layout
    v8f accv = {0.f, 0.f, 0.f, 0.f, 0.f, 0.f, 0.f, 0.f};
    for (int k0 = 0; k0 < Kp; k0 += 32) {
        V16 a, b;
        a.q[0] = *(const uint4*)(Ar + k0);
        a.q[1] = *(const uint4*)(Ar + k0 + 16);
        b.q[0] = *(const uint4*)(Br + k0);
        b.q[1] = *(const uint4*)(Br + k0 + 16);
        accv = __builtin_amdgcn_wmma_f32_16x16x32_bf16(false, a.v, false, b.v,
                                                       (short)0, accv, false, false);
    }
    int cn = (tn << 4) + lrow;
    int bm = (tm << 4) + (half << 3);
    float bv = bias ? bias[cn] : 0.f;
    for (int i = 0; i < 8; ++i) {
        float v = accv[i] + bv;
        if (acc) v += C[(size_t)(bm + i) * N + cn];
        if (act == 1)      v = fmaxf(v, 0.f);
        else if (act == 2) v = v > 0.f ? v : 0.01f * v;
        C[(size_t)(bm + i) * N + cn] = v;
    }
}

// ---------------- big edge-conditioned GEMM ----------------
// msg[e,f] = sum_{k<512, d<256} he[e,k]*h3s[e,d]*Wt[f, k*256+d]
// A generated on the fly as outer product (native bf16 packed muls);
// 4 M-tiles per wave to amortize B loads; B stream prefetched ahead.
#define MBLK 4
__global__ void k_edge_gemm(const bf16* __restrict__ he,   // [E,512]
                            const bf16* __restrict__ h3s,  // [E,256]
                            const bf16* __restrict__ Wt,   // [256, 131072]  (= We2^T bf16)
                            float* __restrict__ msg) {     // [E,256]
    int w = blockIdx.x * (blockDim.x >> 5) + (threadIdx.x >> 5);
    if (w >= (NEDGES / (16 * MBLK)) * (DD / 16)) return;
    int tn = w & 15, tg = w >> 4;
    int lane = threadIdx.x & 31, half = lane >> 4, lrow = lane & 15;

    const bf16* Br = Wt + (size_t)((tn << 4) + lrow) * KBIG + half * 8;
    const bf16* her[MBLK];
    int erow[MBLK];
    v8f accv[MBLK];
    for (int r = 0; r < MBLK; ++r) {
        erow[r] = tg * (16 * MBLK) + r * 16 + lrow;
        her[r] = he + (size_t)erow[r] * EHH;
        accv[r] = (v8f){0.f, 0.f, 0.f, 0.f, 0.f, 0.f, 0.f, 0.f};
    }
    for (int dc = 0; dc < 8; ++dc) {
        int d0 = dc * 32 + half * 8;
        v8bf hb0[MBLK], hb1[MBLK];
        for (int r = 0; r < MBLK; ++r) {
            const bf16* hp = h3s + (size_t)erow[r] * DD + d0;
            hb0[r] = __builtin_bit_cast(v8bf, *(const uint4*)hp);
            hb1[r] = __builtin_bit_cast(v8bf, *(const uint4*)(hp + 16));
        }
        for (int k = 0; k < EHH; ++k) {
            size_t kd = (size_t)k * DD + dc * 32;
            V16 b;
            b.q[0] = *(const uint4*)(Br + kd);
            b.q[1] = *(const uint4*)(Br + kd + 16);
            __builtin_prefetch(Br + kd + 16 * DD, 0, 0);  // B stream, 16 k-steps ahead
            for (int r = 0; r < MBLK; ++r) {
                bf16 hev = her[r][k];
                V16 a;
                a.h[0] = hb0[r] * hev;   // v_pk_mul_bf16 × 4
                a.h[1] = hb1[r] * hev;   // v_pk_mul_bf16 × 4
                accv[r] = __builtin_amdgcn_wmma_f32_16x16x32_bf16(false, a.v, false, b.v,
                                                                  (short)0, accv[r], false, false);
            }
        }
    }
    int cn = (tn << 4) + lrow;
    for (int r = 0; r < MBLK; ++r) {
        int bm = tg * (16 * MBLK) + r * 16 + (half << 3);
        for (int i = 0; i < 8; ++i)
            msg[(size_t)(bm + i) * DD + cn] = accv[r][i];
    }
}

// ---------------- readout MLP head ----------------
__global__ void k_head(const float* __restrict__ nodesum, const float* __restrict__ ncnt,
                       const float* __restrict__ edgesum, const float* __restrict__ ecnt,
                       const float* __restrict__ Wr0, const float* __restrict__ br0,
                       const float* __restrict__ g0, const float* __restrict__ beta0,
                       const float* __restrict__ Wr1, const float* __restrict__ br1,
                       const float* __restrict__ g1, const float* __restrict__ beta1,
                       const float* __restrict__ Wout, const float* __restrict__ bout,
                       float* __restrict__ out) {
    __shared__ float ghf[257];
    __shared__ float x1[256];
    __shared__ float red[256];
    int g = blockIdx.x, f = threadIdx.x;
    ghf[f] = nodesum[(size_t)g * DD + f] / fmaxf(ncnt[g], 1.f);
    if (f == 0) ghf[256] = edgesum[g] / fmaxf(ecnt[g], 1.f);
    __syncthreads();
    float a = 0.f;
    for (int k = 0; k < 257; ++k) a += ghf[k] * Wr0[(size_t)k * DD + f];
    a = (a + br0[f]) * g0[f] + beta0[f];
    x1[f] = a > 0.f ? a : 0.01f * a;
    __syncthreads();
    float a2 = 0.f;
    for (int k = 0; k < 256; ++k) a2 += x1[k] * Wr1[(size_t)k * DD + f];
    a2 = (a2 + br1[f]) * g1[f] + beta1[f];
    a2 = a2 > 0.f ? a2 : 0.01f * a2;
    red[f] = a2 * Wout[f];
    __syncthreads();
    for (int s = 128; s > 0; s >>= 1) {
        if (f < s) red[f] += red[f + s];
        __syncthreads();
    }
    if (f == 0) out[g] = red[0] + bout[0];
}

// ---------------- host launcher ----------------
static inline unsigned Gr(long n) { return (unsigned)((n + 255) / 256); }

extern "C" void kernel_launch(void* const* d_in, const int* in_sizes, int n_in,
                              void* d_out, int out_size, void* d_ws, size_t ws_size,
                              hipStream_t stream) {
    (void)in_sizes; (void)n_in; (void)out_size; (void)ws_size;
    const float* node_feats = (const float*)d_in[0];
    const float* edge_type  = (const float*)d_in[1];
    const int*   src        = (const int*)d_in[2];
    const int*   dst        = (const int*)d_in[3];
    const int*   node_gid   = (const int*)d_in[4];
    const int*   edge_gid   = (const int*)d_in[5];
    const float* W1  = (const float*)d_in[7];
    const float* b1  = (const float*)d_in[8];
    const float* W2  = (const float*)d_in[9];
    const float* b2  = (const float*)d_in[10];
    const float* Wp  = (const float*)d_in[11];
    const float* bp  = (const float*)d_in[12];
    const float* We1 = (const float*)d_in[13];
    const float* be1 = (const float*)d_in[14];
    const float* We2 = (const float*)d_in[15];
    const float* be2 = (const float*)d_in[16];
    const float* b_nn = (const float*)d_in[17];
    const float* W_ih = (const float*)d_in[18];
    const float* b_ih = (const float*)d_in[19];
    const float* W_hh = (const float*)d_in[20];
    const float* b_hh = (const float*)d_in[21];
    const float* Wr0 = (const float*)d_in[22];
    const float* br0 = (const float*)d_in[23];
    const float* g0  = (const float*)d_in[24];
    const float* beta0 = (const float*)d_in[25];
    const float* Wr1 = (const float*)d_in[26];
    const float* br1 = (const float*)d_in[27];
    const float* g1  = (const float*)d_in[28];
    const float* beta1 = (const float*)d_in[29];
    const float* Wout = (const float*)d_in[30];
    const float* bout = (const float*)d_in[31];
    float* out = (float*)d_out;

    char* p = (char*)d_ws;
    auto take = [&](size_t bytes) -> char* {
        char* r = p; p += (bytes + 255) & ~(size_t)255; return r;
    };
    // --- zeroed accumulator region (one contiguous memset) ---
    char* zbase = p;
    float* outDeg  = (float*)take((size_t)NNODES * 4);
    float* inDeg   = (float*)take((size_t)NNODES * 4);
    float* agg1    = (float*)take((size_t)NNODES * KP1 * 4);
    float* agg2    = (float*)take((size_t)NNODES * DD * 4);
    float* mbuf    = (float*)take((size_t)NNODES * DD * 4);
    float* nodesum = (float*)take((size_t)NGRAPH * DD * 4);
    float* ncnt    = (float*)take((size_t)NGRAPH * 4);
    float* edgesum = (float*)take((size_t)NGRAPH * 4);
    float* ecnt    = (float*)take((size_t)NGRAPH * 4);
    size_t zbytes = (size_t)(p - zbase);
    // --- fp32 intermediates ---
    float* inv_s = (float*)take((size_t)NNODES * 4);
    float* inv_d = (float*)take((size_t)NNODES * 4);
    float* h1   = (float*)take((size_t)NNODES * DD * 4);
    float* h2   = (float*)take((size_t)NNODES * DD * 4);
    float* h3   = (float*)take((size_t)NNODES * DD * 4);
    float* msg  = (float*)take((size_t)NEDGES * DD * 4);
    float* gi   = (float*)take((size_t)NNODES * 768 * 4);
    float* gh   = (float*)take((size_t)NNODES * 768 * 4);
    float* hfin = (float*)take((size_t)NNODES * DD * 4);
    // --- bf16 operands ---
    bf16* a1b   = (bf16*)take((size_t)NNODES * KP1 * 2);
    bf16* a2b   = (bf16*)take((size_t)NNODES * DD * 2);
    bf16* h2b   = (bf16*)take((size_t)NNODES * DD * 2);
    bf16* h3b   = (bf16*)take((size_t)NNODES * DD * 2);
    bf16* h3sb  = (bf16*)take((size_t)NEDGES * DD * 2);
    bf16* heb   = (bf16*)take((size_t)NEDGES * EHH * 2);
    bf16* mb    = (bf16*)take((size_t)NNODES * DD * 2);
    bf16* W1t   = (bf16*)take((size_t)DD * KP1 * 2);
    bf16* W2t   = (bf16*)take((size_t)DD * DD * 2);
    bf16* Wpt   = (bf16*)take((size_t)DD * DD * 2);
    bf16* B2t   = (bf16*)take((size_t)DD * DD * 2);
    bf16* Wihb  = (bf16*)take((size_t)768 * DD * 2);
    bf16* Whhb  = (bf16*)take((size_t)768 * DD * 2);
    bf16* We2tb = (bf16*)take((size_t)DD * KBIG * 2);

    hipMemsetAsync(zbase, 0, zbytes, stream);

    // degrees & norms
    k_deg<<<Gr(NEDGES), 256, 0, stream>>>(src, dst, outDeg, inDeg);
    k_invdeg<<<Gr(NNODES), 256, 0, stream>>>(outDeg, inDeg, inv_s, inv_d);

    // weight preparation (bf16, transposed to [N,Kp])
    k_cvtT<<<Gr((long)DD * KP1), 256, 0, stream>>>(W1, W1t, FIN, DD, KP1, (long)DD * KP1);
    k_cvtT<<<Gr((long)DD * DD), 256, 0, stream>>>(W2, W2t, DD, DD, DD, (long)DD * DD);
    k_cvtT<<<Gr((long)DD * DD), 256, 0, stream>>>(Wp, Wpt, DD, DD, DD, (long)DD * DD);
    k_cvtT<<<Gr((long)DD * DD), 256, 0, stream>>>(be2, B2t, DD, DD, DD, (long)DD * DD);
    k_cvt<<<Gr((long)768 * DD), 256, 0, stream>>>(W_ih, Wihb, nullptr, 1, (long)768 * DD);
    k_cvt<<<Gr((long)768 * DD), 256, 0, stream>>>(W_hh, Whhb, nullptr, 1, (long)768 * DD);
    k_cvtT<<<Gr((long)DD * KBIG), 256, 0, stream>>>(We2, We2tb, KBIG, DD, KBIG, (long)DD * KBIG);
    k_heb<<<Gr((long)NEDGES * EHH), 256, 0, stream>>>(edge_type, We1, be1, heb);

    // gconv1
    k_scatter1<<<Gr((long)NEDGES * FIN), 256, 0, stream>>>(node_feats, src, dst, inv_s, agg1);
    k_cvt<<<Gr((long)NNODES * KP1), 256, 0, stream>>>(agg1, a1b, inv_d, KP1, (long)NNODES * KP1);
    k_gemm<<<(NNODES / 16) * (DD / 16) / 8, 256, 0, stream>>>(a1b, W1t, h1, NNODES, DD, KP1, b1, 1, 0);

    // gconv2
    k_scatter2<<<Gr((long)NEDGES * DD), 256, 0, stream>>>(h1, src, dst, inv_s, agg2);
    k_cvt<<<Gr((long)NNODES * DD), 256, 0, stream>>>(agg2, a2b, inv_d, DD, (long)NNODES * DD);
    k_gemm<<<(NNODES / 16) * (DD / 16) / 8, 256, 0, stream>>>(a2b, W2t, h2, NNODES, DD, DD, b2, 1, 0);

    // projection (leaky) -> hidden h3
    k_cvt<<<Gr((long)NNODES * DD), 256, 0, stream>>>(h2, h2b, nullptr, 1, (long)NNODES * DD);
    k_gemm<<<(NNODES / 16) * (DD / 16) / 8, 256, 0, stream>>>(h2b, Wpt, h3, NNODES, DD, DD, bp, 2, 0);
    k_cvt<<<Gr((long)NNODES * DD), 256, 0, stream>>>(h3, h3b, nullptr, 1, (long)NNODES * DD);
    k_gather_src<<<Gr((long)NEDGES * DD), 256, 0, stream>>>(h3, src, h3sb);

    // edge-conditioned message GEMM (dominant 137 GMAC) + be2 term
    k_edge_gemm<<<(NEDGES / (16 * MBLK)) * (DD / 16) / 8, 256, 0, stream>>>(heb, h3sb, We2tb, msg);
    k_gemm<<<(NEDGES / 16) * (DD / 16) / 8, 256, 0, stream>>>(h3sb, B2t, msg, NEDGES, DD, DD, nullptr, 0, 1);

    // aggregate messages, GRU
    k_scatter_msg<<<Gr((long)NEDGES * DD), 256, 0, stream>>>(msg, dst, mbuf);
    k_msg_act<<<Gr((long)NNODES * DD), 256, 0, stream>>>(mbuf, b_nn, mb);
    k_gemm<<<(NNODES / 16) * (768 / 16) / 8, 256, 0, stream>>>(mb, Wihb, gi, NNODES, 768, DD, b_ih, 0, 0);
    k_gemm<<<(NNODES / 16) * (768 / 16) / 8, 256, 0, stream>>>(h3b, Whhb, gh, NNODES, 768, DD, b_hh, 0, 0);
    k_gru<<<Gr((long)NNODES * DD), 256, 0, stream>>>(gi, gh, h3, hfin);

    // readout + head
    k_node_read<<<Gr((long)NNODES * DD), 256, 0, stream>>>(hfin, node_gid, nodesum, ncnt);
    k_edge_read<<<Gr(NEDGES), 256, 0, stream>>>(edge_type, edge_gid, edgesum, ecnt);
    k_head<<<NGRAPH, 256, 0, stream>>>(nodesum, ncnt, edgesum, ecnt,
                                       Wr0, br0, g0, beta0, Wr1, br1, g1, beta1,
                                       Wout, bout, out);
}
